// ImplicitWarpModule_31696858644753
// MI455X (gfx1250) — matrix operand order repaired and checked by reading, not compiled, not run on hardware
//
#include <hip/hip_runtime.h>
#include <hip/hip_bf16.h>
#include <math.h>

typedef _Float16 f16;
typedef __attribute__((ext_vector_type(4)))  _Float16 v4h;
typedef __attribute__((ext_vector_type(8)))  _Float16 v8h;
typedef __attribute__((ext_vector_type(16))) _Float16 v16h;
typedef __attribute__((ext_vector_type(8)))  float    v8f;

#define NWAVES  2
#define THREADS (NWAVES * 32)
#define PEDIM   48
#define NPF     24
#define LN1E4   9.210340371976184f
#define TWOPI   6.283185307179586f

// ---- A-fragment: 16-bit A 16x32 layout; tokens row-major [row][stride] ----
// lane: M = l&15 ; elems 0..7 -> K = kc*32 + hi*8 + 0..7 ; elems 8..15 -> +16
__device__ __forceinline__ v16h load_a_frag(const f16* tok, int row, int hi, int kc, int stride) {
  const f16* p = tok + row * stride + kc * 32 + hi * 8;
  v8h lo = *(const v8h*)p;
  v8h hh = *(const v8h*)(p + 16);
  return __builtin_shufflevector(lo, hh, 0, 1, 2, 3, 4, 5, 6, 7,
                                         8, 9, 10, 11, 12, 13, 14, 15);
}

// ---- B-fragment: 16-bit B 32x16 layout; weights N-major [n][stride] ----
// lane: N = l&15 ; elems 0..7 -> K = kc*32 + hi*8 + 0..7 ; elems 8..15 -> +16
__device__ __forceinline__ v16h load_b_frag(const f16* wt, int nt, int kc, int lm, int hi, int stride) {
  const f16* p = wt + (nt * 16 + lm) * stride + kc * 32 + hi * 8;
  v8h lo = *(const v8h*)p;
  v8h hh = *(const v8h*)(p + 16);
  return __builtin_shufflevector(lo, hh, 0, 1, 2, 3, 4, 5, 6, 7,
                                         8, 9, 10, 11, 12, 13, 14, 15);
}

// ---- C store: results column-major [col][rows]; 8 rows of a lane contiguous ----
__device__ __forceinline__ void store_c(f16* resT, int stride, int col, int rowBase, const v8f acc) {
  v8h h;
#pragma unroll
  for (int r = 0; r < 8; ++r) h[r] = (f16)acc[r];
  *(v8h*)(resT + col * stride + rowBase) = h;
}

__global__ __launch_bounds__(THREADS)
void implicit_warp_kernel(const float* __restrict__ y, const float* __restrict__ x,
                          const float* __restrict__ flow,
                          const float* __restrict__ Wq, const float* __restrict__ bq,
                          const float* __restrict__ Wk, const float* __restrict__ bk,
                          const float* __restrict__ Wv, const float* __restrict__ bv,
                          float* __restrict__ out, int H, int W, int P) {
  // Folded weights (N-major, zero padded):
  //  sWqp [64 n][64 k]: k<48 -> Wq[n,k] ; 48..50 -> channel-fold WqC ; 51 -> bq ; rest 0
  //  sWkp/sWvp [64 n][32 k]: 0..2 -> channel-fold ; 3..6 -> winb[v]@W^T + b ; rest 0
  __shared__ f16   sWqp[64 * 64];
  __shared__ f16   sWkp[64 * 32];
  __shared__ f16   sWvp[64 * 32];
  __shared__ float sWinB[4][PEDIM];
  __shared__ float sInvDt[NPF];
  __shared__ f16   sWave[NWAVES][9216];   // tokens (3072) then q/k/v results (9216)

  const int tid  = threadIdx.x;
  const int wid  = tid >> 5;
  const int lane = tid & 31;
  const int lm   = lane & 15;
  const int hi   = lane >> 4;
  const int HW   = H * W;

  // ---------- Phase 0a: PE tables ----------
  if (tid < NPF)
    sInvDt[tid] = __expf(-((float)(2 * (tid >> 1)) / 24.0f) * LN1E4);
  for (int e = tid; e < 4 * PEDIM; e += THREADS) {
    int v = e / PEDIM, k = e - v * PEDIM;
    int kk = (k < NPF) ? k : (k - NPF);
    float base = (k < NPF) ? (float)(v >> 1) : (float)(v & 1);
    float invd = __expf(-((float)(2 * (kk >> 1)) / 24.0f) * LN1E4);
    float t = base * (TWOPI / (1.0f + 1e-6f)) * invd;
    sWinB[v][k] = (kk & 1) ? __cosf(t) : __sinf(t);
  }
  __syncthreads();

  // ---------- Phase 0b: folded weights ----------
  for (int e = tid; e < 64 * 64; e += THREADS) {
    int n = e >> 6, k = e & 63;
    float val = 0.0f;
    if (k < PEDIM) {
      val = Wq[n * PEDIM + k];
    } else if (k < 51) {
      int c = k - 48; float s = 0.0f;
      for (int j = 0; j < 16; ++j) s += Wq[n * PEDIM + 3 * j + c];
      val = s;
    } else if (k == 51) {
      val = bq[n];
    }
    sWqp[n * 64 + k] = (f16)val;
  }
  for (int e = tid; e < 2 * 64 * 32; e += THREADS) {
    int m = e >> 11;
    int rem = e & 2047;
    int n = rem >> 5, k = rem & 31;
    const float* Wsrc = m ? Wv : Wk;
    const float* bsrc = m ? bv : bk;
    float val = 0.0f;
    if (k < 3) {
      float s = 0.0f;
      for (int j = 0; j < 16; ++j) s += Wsrc[n * PEDIM + 3 * j + k];
      val = s;
    } else if (k < 7) {
      int v = k - 3; float s = bsrc[n];
      for (int kk = 0; kk < PEDIM; ++kk) s += sWinB[v][kk] * Wsrc[n * PEDIM + kk];
      val = s;
    }
    f16* dst = m ? sWvp : sWkp;
    dst[n * 32 + k] = (f16)val;
  }

  // ---------- Phase 1: build tokens ----------
  const int pb = (blockIdx.x * NWAVES + wid) * 16;   // first pixel of this wave
  f16* tok   = &sWave[wid][0];            // q tokens  [16][64]
  f16* kvtok = tok + 16 * 64;             // kv tokens [64][32]

  if (lane < 16) {                        // one query token per lane
    int gp = pb + lane; if (gp >= P) gp = P - 1;
    int ni = gp / HW, sp = gp - ni * HW;
    int gy = sp / W,  gx = sp - gy * W;
    float f0 = flow[((size_t)ni * HW + sp) * 2 + 0];
    float f1 = flow[((size_t)ni * HW + sp) * 2 + 1];
    float wy = (float)gy + f1, wx = (float)gx + f0;
    float argy = (wy - floorf(wy)) * (TWOPI / (2.0f + 1e-6f));
    float argx = (wx - floorf(wx)) * (TWOPI / (2.0f + 1e-6f));
    float xc[3];
#pragma unroll
    for (int c = 0; c < 3; ++c) xc[c] = x[(size_t)(ni * 3 + c) * HW + sp];
#pragma unroll
    for (int c = 0; c < 6; ++c) {        // k = 0..47 : subpixel-offset PE
      v8h hh;
#pragma unroll
      for (int i = 0; i < 8; ++i) {
        int k = c * 8 + i;
        int kk = (k < NPF) ? k : (k - NPF);
        float arg = ((k < NPF) ? argy : argx) * sInvDt[kk];
        hh[i] = (f16)((kk & 1) ? __cosf(arg) : __sinf(arg));
      }
      *(v8h*)(tok + lane * 64 + c * 8) = hh;
    }
    {                                     // k = 48..55 : [xc0,xc1,xc2,1,0,0,0,0]
      v8h hh = {};
      hh[0] = (f16)xc[0]; hh[1] = (f16)xc[1]; hh[2] = (f16)xc[2];
      hh[3] = (f16)1.0f;
      *(v8h*)(tok + lane * 64 + 48) = hh;
      v8h zz = {};
      *(v8h*)(tok + lane * 64 + 56) = zz;  // k = 56..63
    }
  }
#pragma unroll
  for (int rr = 0; rr < 2; ++rr) {        // two key/value rows per lane
    int row = lane * 2 + rr;
    int pl = row >> 2, v = row & 3;
    int gp = pb + pl; if (gp >= P) gp = P - 1;
    int ni = gp / HW, sp = gp - ni * HW;
    int gy = sp / W,  gx = sp - gy * W;
    float f0 = flow[((size_t)ni * HW + sp) * 2 + 0];
    float f1 = flow[((size_t)ni * HW + sp) * 2 + 1];
    int iy = (int)floorf((float)gy + f1) + (v >> 1);
    int ix = (int)floorf((float)gx + f0) + (v & 1);
    iy = min(max(iy, 0), H - 1);
    ix = min(max(ix, 0), W - 1);
    v8h h0 = {};
#pragma unroll
    for (int c = 0; c < 3; ++c)
      h0[c] = (f16)y[(size_t)(ni * 3 + c) * HW + (size_t)iy * W + ix];
    h0[3 + v] = (f16)1.0f;                // onehot selects winb[v]@W^T + b row
    v8h zz = {};
    *(v8h*)(kvtok + row * 32 + 0)  = h0;
    *(v8h*)(kvtok + row * 32 + 8)  = zz;
    *(v8h*)(kvtok + row * 32 + 16) = zz;
    *(v8h*)(kvtok + row * 32 + 24) = zz;
  }
  __syncthreads();

  // ---------- Phase 2: pull all A-fragments into registers ----------
  v16h qA[2], kvA[4];
#pragma unroll
  for (int kc = 0; kc < 2; ++kc) qA[kc] = load_a_frag(tok, lm, hi, kc, 64);
#pragma unroll
  for (int mt = 0; mt < 4; ++mt) kvA[mt] = load_a_frag(kvtok, mt * 16 + lm, hi, 0, 32);

  // ---------- Phase 3: WMMA projections (staging reused, column-major results) ----------
  f16* qresT = &sWave[wid][0];        // [64 col][16 row]
  f16* kresT = qresT + 64 * 16;       // [64 col][64 row]
  f16* vresT = kresT + 64 * 64;       // [64 col][64 row]

  for (int nt = 0; nt < 4; ++nt) {
    int col = nt * 16 + lm;
    {   // Q : K = 64 (PE 48 + channels + bias), 2 WMMA steps
      v16h b0 = load_b_frag(sWqp, nt, 0, lm, hi, 64);
      v16h b1 = load_b_frag(sWqp, nt, 1, lm, hi, 64);
      v8f acc = {};
      acc = __builtin_amdgcn_wmma_f32_16x16x32_f16(false, qA[0], false, b0, (short)0, acc, false, false);
      acc = __builtin_amdgcn_wmma_f32_16x16x32_f16(false, qA[1], false, b1, (short)0, acc, false, false);
      store_c(qresT, 16, col, hi * 8, acc);
    }
    {   // K : folded K = 32, single WMMA step per tile
      v16h b0 = load_b_frag(sWkp, nt, 0, lm, hi, 32);
#pragma unroll
      for (int mt = 0; mt < 4; ++mt) {
        v8f acc = {};
        acc = __builtin_amdgcn_wmma_f32_16x16x32_f16(false, kvA[mt], false, b0, (short)0, acc, false, false);
        store_c(kresT, 64, col, mt * 16 + hi * 8, acc);
      }
    }
    {   // V : folded K = 32
      v16h b0 = load_b_frag(sWvp, nt, 0, lm, hi, 32);
#pragma unroll
      for (int mt = 0; mt < 4; ++mt) {
        v8f acc = {};
        acc = __builtin_amdgcn_wmma_f32_16x16x32_f16(false, kvA[mt], false, b0, (short)0, acc, false, false);
        store_c(vresT, 64, col, mt * 16 + hi * 8, acc);
      }
    }
  }

  // ---------- Phase 4: per-pixel, per-head attention over nv=4 ----------
  const float scale = 0.3535533905932738f;   // head_dim^-0.5, head_dim = 8
#pragma unroll
  for (int t = 0; t < 4; ++t) {
    int pair = lane + 32 * t;    // 128 (pixel,head) pairs per wave
    int px = pair >> 3, hd = pair & 7;
    float q8[8];
#pragma unroll
    for (int d = 0; d < 8; ++d) q8[d] = (float)qresT[(hd * 8 + d) * 16 + px] * scale;
    float s0 = 0.f, s1 = 0.f, s2 = 0.f, s3 = 0.f;
#pragma unroll
    for (int d = 0; d < 8; ++d) {
      v4h kd = *(const v4h*)(kresT + (hd * 8 + d) * 64 + px * 4);
      s0 += q8[d] * (float)kd[0];
      s1 += q8[d] * (float)kd[1];
      s2 += q8[d] * (float)kd[2];
      s3 += q8[d] * (float)kd[3];
    }
    float mx = fmaxf(fmaxf(s0, s1), fmaxf(s2, s3));
    float e0 = __expf(s0 - mx), e1 = __expf(s1 - mx);
    float e2 = __expf(s2 - mx), e3 = __expf(s3 - mx);
    float inv = 1.0f / (e0 + e1 + e2 + e3);
    int gp = pb + px;
    if (gp < P) {
      int ni = gp / HW, sp = gp - ni * HW;
#pragma unroll
      for (int d = 0; d < 8; ++d) {
        v4h vd = *(const v4h*)(vresT + (hd * 8 + d) * 64 + px * 4);
        float o = e0 * (float)vd[0] + e1 * (float)vd[1] +
                  e2 * (float)vd[2] + e3 * (float)vd[3];
        out[((size_t)(ni * 64 + hd * 8 + d)) * HW + sp] = o * inv;
      }
    }
  }
}

extern "C" void kernel_launch(void* const* d_in, const int* in_sizes, int n_in,
                              void* d_out, int out_size, void* d_ws, size_t ws_size,
                              hipStream_t stream) {
  (void)n_in; (void)out_size; (void)d_ws; (void)ws_size;
  const float* y    = (const float*)d_in[0];
  const float* x    = (const float*)d_in[1];
  const float* flow = (const float*)d_in[2];
  const float* Wq   = (const float*)d_in[3];
  const float* bq   = (const float*)d_in[4];
  const float* Wk   = (const float*)d_in[5];
  const float* bk   = (const float*)d_in[6];
  const float* Wv   = (const float*)d_in[7];
  const float* bv   = (const float*)d_in[8];
  float* out = (float*)d_out;

  const int H = 256, W = 256;
  const int P = in_sizes[2] / 2;                 // n*h*w from flow size
  const int blocks = (P + NWAVES * 16 - 1) / (NWAVES * 16);
  implicit_warp_kernel<<<blocks, THREADS, 0, stream>>>(y, x, flow, Wq, bq, Wk, bk, Wv, bv,
                                                       out, H, W, P);
}
